// LearnedChambollePock_73701638799427
// MI455X (gfx1250) — matrix operand order repaired and checked by reading, not compiled, not run on hardware
//
#include <hip/hip_runtime.h>
#include <hip/hip_bf16.h>

typedef __attribute__((ext_vector_type(16))) _Float16 v16h;
typedef __attribute__((ext_vector_type(8)))  _Float16 h8;
typedef __attribute__((ext_vector_type(8)))  float    v8f;

#define HH_IMG 128
#define WW_IMG 128
#define A_ANG  96
#define D_DET  128
#define T_SAMP 182
#define BATCH  4
#define OPNORM 128.0f

// ---------------------------------------------------------------------------
// small elementwise kernels
// ---------------------------------------------------------------------------
__global__ __launch_bounds__(256) void zero_k(float* p, int n) {
    int i = blockIdx.x * blockDim.x + threadIdx.x;
    if (i < n) p[i] = 0.0f;
}

__global__ void trig_k(float* cosv, float* sinv) {
    int a = threadIdx.x;
    if (a < A_ANG) {
        float ang = (float)a * (3.14159265358979323846f / (float)A_ANG);
        cosv[a] = cosf(ang);
        sinv[a] = sinf(ang);
    }
}

__global__ __launch_bounds__(256) void dual_update_k(float* dual, const float* o5,
                                                     const float* sigma, int n) {
    int i = blockIdx.x * blockDim.x + threadIdx.x;
    if (i < n) dual[i] += sigma[0] * o5[i];
}

__global__ __launch_bounds__(256) void primal_update_k(float* primal, float* pbar,
                                                       const float* o5,
                                                       const float* tau, const float* theta,
                                                       int n) {
    int i = blockIdx.x * blockDim.x + threadIdx.x;
    if (i < n) {
        float p  = primal[i];
        float pn = p + tau[0] * o5[i];
        primal[i] = pn;
        pbar[i]   = pn + theta[0] * (pn - p);
    }
}

__global__ __launch_bounds__(256) void extract_k(const float* primal, float* out, int n) {
    int i = blockIdx.x * blockDim.x + threadIdx.x;
    if (i < n) out[i] = primal[i * 5 + 0];
}

// ---------------------------------------------------------------------------
// Radon forward: sino[b,a,d] = sum_t bilinear(img_ch1, rot(s,t)) / OPNORM
// ---------------------------------------------------------------------------
__global__ __launch_bounds__(128) void radon_fwd_k(const float* __restrict__ pbar,
                                                   const float* __restrict__ cosv,
                                                   const float* __restrict__ sinv,
                                                   float* __restrict__ sino) {
    int idx = blockIdx.x * blockDim.x + threadIdx.x;
    if (idx >= BATCH * A_ANG * D_DET) return;
    int d = idx & (D_DET - 1);
    int a = (idx >> 7) % A_ANG;
    int b = idx / (A_ANG * D_DET);

    float s  = (float)d - (float)(D_DET - 1) * 0.5f;
    float c  = cosv[a];
    float sn = sinv[a];
    const float* img = pbar + (size_t)b * HH_IMG * WW_IMG * 5;

    float acc = 0.0f;
    for (int t = 0; t < T_SAMP; ++t) {
        float tt = (float)t - (float)(T_SAMP - 1) * 0.5f;
        float xi = s * c - tt * sn + (float)(WW_IMG - 1) * 0.5f;
        float yi = s * sn + tt * c + (float)(HH_IMG - 1) * 0.5f;
        float xf = floorf(xi), yf = floorf(yi);
        float wx = xi - xf, wy = yi - yf;
        int x0 = (int)xf, y0 = (int)yf;
        int x1 = x0 + 1, y1 = y0 + 1;

        float v00 = 0.f, v01 = 0.f, v10 = 0.f, v11 = 0.f;
        bool xv0 = (x0 >= 0) & (x0 < WW_IMG);
        bool xv1 = (x1 >= 0) & (x1 < WW_IMG);
        bool yv0 = (y0 >= 0) & (y0 < HH_IMG);
        bool yv1 = (y1 >= 0) & (y1 < HH_IMG);
        if (yv0 & xv0) v00 = img[(y0 * WW_IMG + x0) * 5 + 1];
        if (yv0 & xv1) v01 = img[(y0 * WW_IMG + x1) * 5 + 1];
        if (yv1 & xv0) v10 = img[(y1 * WW_IMG + x0) * 5 + 1];
        if (yv1 & xv1) v11 = img[(y1 * WW_IMG + x1) * 5 + 1];
        acc += v00 * (1.f - wy) * (1.f - wx) + v01 * (1.f - wy) * wx
             + v10 * wy * (1.f - wx)        + v11 * wy * wx;
    }
    sino[idx] = acc * (1.0f / OPNORM);
}

// ---------------------------------------------------------------------------
// Radon adjoint
// ---------------------------------------------------------------------------
__global__ __launch_bounds__(128) void radon_adj_k(const float* __restrict__ dual,
                                                   const float* __restrict__ cosv,
                                                   const float* __restrict__ sinv,
                                                   float* __restrict__ adj) {
    int idx = blockIdx.x * blockDim.x + threadIdx.x;
    if (idx >= BATCH * HH_IMG * WW_IMG) return;
    int x = idx & (WW_IMG - 1);
    int y = (idx >> 7) & (HH_IMG - 1);
    int b = idx >> 14;

    float px = (float)x - (float)(WW_IMG - 1) * 0.5f;
    float py = (float)y - (float)(HH_IMG - 1) * 0.5f;
    const float* sb = dual + (size_t)b * A_ANG * D_DET * 5;

    float acc = 0.0f;
    for (int a = 0; a < A_ANG; ++a) {
        float det = px * cosv[a] + py * sinv[a] + (float)(D_DET - 1) * 0.5f;
        float d0f = floorf(det);
        float w   = det - d0f;
        int d0 = (int)d0f;
        int d1 = d0 + 1;
        float v0 = (d0 >= 0 && d0 < D_DET) ? sb[(a * D_DET + d0) * 5 + 0] : 0.f;
        float v1 = (d1 >= 0 && d1 < D_DET) ? sb[(a * D_DET + d1) * 5 + 0] : 0.f;
        acc += v0 * (1.f - w) + v1 * w;
    }
    adj[idx] = acc * (1.0f / OPNORM);
}

// ---------------------------------------------------------------------------
// 3x3 SAME conv, implicit GEMM on V_WMMA_F32_16X16X32_F16.
//
// LDS holds an explicit zero-padded im2col panel  sAi[PIXWG][KPAD]  (f16,
// K-contiguous) and a transposed weight panel     sWt[CPAD ][KPAD]  (f16).
// With these layouts each WMMA operand fragment is exactly two contiguous
// 16-byte runs per lane -> 2x ds_load_b128 per fragment, no guards, EXEC
// stays all-ones through the WMMA loop.
//
// MODE 0: plain NHWC tensor (CIN channels) in x0p
// MODE 1: dual concat   : x0p=dual(5ch), x1p=evalop(1ch), x2p=y(1ch)  (CIN=7)
// MODE 2: primal concat : x0p=primal(5ch), x1p=adj(1ch)               (CIN=6)
// ---------------------------------------------------------------------------
template<int CIN, int COUT, int MODE, bool RELU>
__global__ __launch_bounds__(128)
void conv3x3_wmma_k(const float* __restrict__ x0p,
                    const float* __restrict__ x1p,
                    const float* __restrict__ x2p,
                    const float* __restrict__ wp,   // HWIO [3,3,CIN,COUT]
                    const float* __restrict__ bp,   // [COUT]
                    float* __restrict__ outp,       // NHWC [B,HH,WW,COUT]
                    int HH, int WW) {
    constexpr int NT     = (COUT + 15) / 16;   // out-channel tiles (1 or 2)
    constexpr int XT     = 4 / NT;             // pixel tiles per WG
    constexpr int PIXWG  = 16 * XT;            // pixels per WG (32 or 64)
    constexpr int KTOT   = 9 * CIN;
    constexpr int KSTEPS = (KTOT + 31) / 32;
    constexpr int KPAD   = KSTEPS * 32;
    constexpr int CPAD   = 16 * NT;

    __shared__ __align__(16) _Float16 sAi[PIXWG * KPAD];  // im2col panel
    __shared__ __align__(16) _Float16 sWt[CPAD * KPAD];   // weights, co-major

    const int tid = threadIdx.x;
    const int b   = blockIdx.z;
    const int yy  = blockIdx.y;
    const int x0  = blockIdx.x * PIXWG;

    // helper: im2col element (pixel m, k) -> f32 value (zero padded)
    auto fetchA = [&](int m, int k) -> float {
        float v = 0.0f;
        if (k < KTOT) {
            int tap = k / CIN;              // constexpr divisor -> mul/shift
            int c   = k - tap * CIN;
            int ky  = tap / 3;
            int kx  = tap - ky * 3;
            int gy  = yy + ky - 1;
            int gx  = x0 + m + kx - 1;
            if (gy >= 0 && gy < HH && gx >= 0 && gx < WW) {
                int p = (b * HH + gy) * WW + gx;
                if (MODE == 0)      v = x0p[p * CIN + c];
                else if (MODE == 1) v = (c < 5) ? x0p[p * 5 + c] : ((c == 5) ? x1p[p] : x2p[p]);
                else                v = (c < 5) ? x0p[p * 5 + c] : x1p[p];
            }
        }
        return v;
    };

    // --- build im2col activation panel ---
    if constexpr ((CIN & 1) == 0) {
        // even CIN: k, k+1 never straddle a tap -> packed 32-bit LDS stores
        unsigned* sAu = (unsigned*)sAi;
        for (int idx = tid; idx < (PIXWG * KPAD) / 2; idx += 128) {
            int m = (2 * idx) / KPAD;
            int k = (2 * idx) - m * KPAD;
            union { _Float16 h[2]; unsigned u; } pk;
            pk.h[0] = (_Float16)fetchA(m, k);
            pk.h[1] = (_Float16)fetchA(m, k + 1);
            sAu[idx] = pk.u;
        }
    } else {
        for (int idx = tid; idx < PIXWG * KPAD; idx += 128) {
            int m = idx / KPAD;
            int k = idx - m * KPAD;
            sAi[idx] = (_Float16)fetchA(m, k);
        }
    }

    // --- build transposed weight panel: sWt[co][k] (zero padded) ---
    {
        unsigned* sWu = (unsigned*)sWt;
        for (int idx = tid; idx < (CPAD * KPAD) / 2; idx += 128) {
            int co = (2 * idx) / KPAD;
            int k  = (2 * idx) - co * KPAD;
            union { _Float16 h[2]; unsigned u; } pk;
            pk.h[0] = (_Float16)((k     < KTOT && co < COUT) ? wp[(k    ) * COUT + co] : 0.0f);
            pk.h[1] = (_Float16)((k + 1 < KTOT && co < COUT) ? wp[(k + 1) * COUT + co] : 0.0f);
            sWu[idx] = pk.u;
        }
    }
    __syncthreads();

    const int wv   = tid >> 5;
    const int lane = tid & 31;
    const int xt   = (NT == 2) ? (wv >> 1) : wv;   // pixel tile of this wave
    const int ct   = (NT == 2) ? (wv & 1) : 0;     // out-channel tile
    const int mr   = lane & 15;                    // A row / B col / D col
    const int g    = lane >> 4;                    // lane-group (K half)

    // per-lane row bases (K-contiguous, 16B aligned at h8 granularity)
    const h8* rowA = (const h8*)(sAi + (size_t)(xt * 16 + mr) * KPAD);
    const h8* rowB = (const h8*)(sWt + (size_t)(ct * 16 + mr) * KPAD);

    v8f acc = {0.f, 0.f, 0.f, 0.f, 0.f, 0.f, 0.f, 0.f};

#pragma unroll
    for (int kk = 0; kk < KSTEPS; ++kk) {
        // A fragment: lane-group g needs K = kk*32 + {8g..8g+7, 16+8g..16+8g+7}
        h8 a0 = rowA[kk * 4 + g];
        h8 a1 = rowA[kk * 4 + 2 + g];
        // B fragment: lane-group g needs K = kk*32 + {8g..8g+7, 16+8g..16+8g+7}
        h8 b0 = rowB[kk * 4 + g];
        h8 b1 = rowB[kk * 4 + 2 + g];
        v16h a  = __builtin_shufflevector(a0, a1, 0,1,2,3,4,5,6,7,8,9,10,11,12,13,14,15);
        v16h bb = __builtin_shufflevector(b0, b1, 0,1,2,3,4,5,6,7,8,9,10,11,12,13,14,15);
        acc = __builtin_amdgcn_wmma_f32_16x16x32_f16(
            /*neg_a=*/false, a, /*neg_b=*/false, bb,
            /*c_mod=*/(short)0, acc, /*reuse_a=*/false, /*reuse_b=*/false);
    }

    // D layout: VGPR r holds M = r + 8*g ; N = lane%16
    const int co = ct * 16 + mr;
    if (co < COUT) {
        float bias = bp[co];
#pragma unroll
        for (int r = 0; r < 8; ++r) {
            int mm = r + (g << 3);
            int px = x0 + xt * 16 + mm;
            float v = acc[r] + bias;
            if (RELU) v = fmaxf(v, 0.0f);
            outp[((b * HH + yy) * WW + px) * COUT + co] = v;
        }
    }
}

// ---------------------------------------------------------------------------
// driver
// ---------------------------------------------------------------------------
extern "C" void kernel_launch(void* const* d_in, const int* in_sizes, int n_in,
                              void* d_out, int out_size, void* d_ws, size_t ws_size,
                              hipStream_t stream) {
    (void)in_sizes; (void)n_in; (void)ws_size;

    const float* y_in  = (const float*)d_in[0];
    const float* sigma = (const float*)d_in[1];
    const float* tau   = (const float*)d_in[2];
    const float* theta = (const float*)d_in[3];
    const float* dw1 = (const float*)d_in[4];   const float* db1 = (const float*)d_in[5];
    const float* dw2 = (const float*)d_in[6];   const float* db2 = (const float*)d_in[7];
    const float* dw3 = (const float*)d_in[8];   const float* db3 = (const float*)d_in[9];
    const float* pw1 = (const float*)d_in[10];  const float* pb1 = (const float*)d_in[11];
    const float* pw2 = (const float*)d_in[12];  const float* pb2 = (const float*)d_in[13];
    const float* pw3 = (const float*)d_in[14];  const float* pb3 = (const float*)d_in[15];
    float* out = (float*)d_out;

    // workspace layout (floats)
    float* ws = (float*)d_ws;
    const size_t N_PRI  = (size_t)BATCH * HH_IMG * WW_IMG * 5;   // 327680
    const size_t N_DUA  = (size_t)BATCH * A_ANG * D_DET * 5;     // 245760
    const size_t N_SIN  = (size_t)BATCH * A_ANG * D_DET;         // 49152
    const size_t N_ADJ  = (size_t)BATCH * HH_IMG * WW_IMG;       // 65536
    const size_t N_T32  = (size_t)BATCH * HH_IMG * WW_IMG * 32;  // 2097152 (covers sino too)

    float* cosv   = ws;                 // 96
    float* sinv   = ws + 128;           // 96
    float* primal = ws + 256;
    float* pbar   = primal + N_PRI;
    float* dual   = pbar + N_PRI;
    float* sino   = dual + N_DUA;
    float* adj    = sino + N_SIN;
    float* t1     = adj + N_ADJ;        // conv hidden 1 / also conv3 output
    float* t2     = t1 + N_T32;         // conv hidden 2

    auto g1 = [](size_t n) { return dim3((unsigned)((n + 255) / 256)); };

    // init
    trig_k<<<1, 128, 0, stream>>>(cosv, sinv);
    zero_k<<<g1(N_PRI), 256, 0, stream>>>(primal, (int)N_PRI);
    zero_k<<<g1(N_PRI), 256, 0, stream>>>(pbar, (int)N_PRI);
    zero_k<<<g1(N_DUA), 256, 0, stream>>>(dual, (int)N_DUA);

    const dim3 blk(128);
    const dim3 gFwd((unsigned)((BATCH * A_ANG * D_DET + 127) / 128));
    const dim3 gAdj((unsigned)((BATCH * HH_IMG * WW_IMG + 127) / 128));
    const dim3 gS32(D_DET / 32, A_ANG, BATCH);    // sino convs, COUT=32
    const dim3 gS64(D_DET / 64, A_ANG, BATCH);    // sino conv3, COUT=5
    const dim3 gI32(WW_IMG / 32, HH_IMG, BATCH);  // image convs, COUT=32
    const dim3 gI64(WW_IMG / 64, HH_IMG, BATCH);  // image conv3, COUT=5

    for (int it = 0; it < 10; ++it) {
        // evalop = radon_fwd(pbar[...,1]) / OPNORM
        radon_fwd_k<<<gFwd, blk, 0, stream>>>(pbar, cosv, sinv, sino);

        // dual block: concat(dual, evalop, y) -> 7->32->32->5
        conv3x3_wmma_k<7, 32, 1, true ><<<gS32, blk, 0, stream>>>(
            dual, sino, y_in, dw1, db1, t1, A_ANG, D_DET);
        conv3x3_wmma_k<32, 32, 0, true ><<<gS32, blk, 0, stream>>>(
            t1, nullptr, nullptr, dw2, db2, t2, A_ANG, D_DET);
        conv3x3_wmma_k<32, 5, 0, false><<<gS64, blk, 0, stream>>>(
            t2, nullptr, nullptr, dw3, db3, t1, A_ANG, D_DET);
        dual_update_k<<<g1(N_DUA), 256, 0, stream>>>(dual, t1, sigma, (int)N_DUA);

        // evalop_adj = radon_adj(dual[...,0]) / OPNORM
        radon_adj_k<<<gAdj, blk, 0, stream>>>(dual, cosv, sinv, adj);

        // primal block: concat(primal, adj) -> 6->32->32->5
        conv3x3_wmma_k<6, 32, 2, true ><<<gI32, blk, 0, stream>>>(
            primal, adj, nullptr, pw1, pb1, t1, HH_IMG, WW_IMG);
        conv3x3_wmma_k<32, 32, 0, true ><<<gI32, blk, 0, stream>>>(
            t1, nullptr, nullptr, pw2, pb2, t2, HH_IMG, WW_IMG);
        conv3x3_wmma_k<32, 5, 0, false><<<gI64, blk, 0, stream>>>(
            t2, nullptr, nullptr, pw3, pb3, t1, HH_IMG, WW_IMG);

        // primal += tau*block ; pbar = (1+theta)*new - theta*old
        primal_update_k<<<g1(N_PRI), 256, 0, stream>>>(primal, pbar, t1, tau, theta, (int)N_PRI);
    }

    // out = primal[..., 0]
    extract_k<<<g1((size_t)out_size), 256, 0, stream>>>(primal, out, out_size);
}
